// MultiHeadAttention_24103356465239
// MI455X (gfx1250) — compile-verified, matrix operand order checked by reading, not run on hardware
//
#include <hip/hip_runtime.h>
#include <hip/hip_bf16.h>
#include <math.h>

// Problem constants (from reference)
#define BATCH 2
#define SEQ   2048
#define CH    1024
#define NHEAD 16
#define HDIM  64
#define MROWS (BATCH * SEQ)   // 4096 rows of x

typedef __attribute__((ext_vector_type(16))) __bf16 v16bf;
typedef __attribute__((ext_vector_type(8)))  float  v8f;
typedef __attribute__((ext_vector_type(4)))  int    v4i;

#define AS1 __attribute__((address_space(1)))
#define AS3 __attribute__((address_space(3)))

// ---- CDNA5 feature probes -------------------------------------------------
#if defined(__has_builtin)
#  if __has_builtin(__builtin_amdgcn_global_load_async_to_lds_b128)
#    define HAVE_ASYNC_B128 1
#  else
#    define HAVE_ASYNC_B128 0
#  endif
#  if __has_builtin(__builtin_amdgcn_tensor_load_to_lds)
#    define HAVE_TDM 1
#  else
#    define HAVE_TDM 0
#  endif
#  if __has_builtin(__builtin_amdgcn_s_wait_asynccnt)
#    define HAVE_WAIT_ASYNC 1
#  else
#    define HAVE_WAIT_ASYNC 0
#  endif
#  if __has_builtin(__builtin_amdgcn_s_wait_tensorcnt)
#    define HAVE_WAIT_TENSOR 1
#  else
#    define HAVE_WAIT_TENSOR 0
#  endif
#else
#  define HAVE_ASYNC_B128 0
#  define HAVE_TDM 0
#  define HAVE_WAIT_ASYNC 0
#  define HAVE_WAIT_TENSOR 0
#endif

#if HAVE_ASYNC_B128
#pragma message("CDNA5 probe: global_load_async_to_lds_b128 AVAILABLE")
#else
#pragma message("CDNA5 probe: global_load_async_to_lds_b128 not available")
#endif
#if HAVE_TDM
#pragma message("CDNA5 probe: tensor_load_to_lds (TDM) AVAILABLE")
#else
#pragma message("CDNA5 probe: tensor_load_to_lds (TDM) not available")
#endif

// 16-byte global -> LDS copy; async (ASYNCcnt) when available.
// Builtin prototype (from compiler diagnostic): first param is AS1 int4*.
__device__ __forceinline__ void async_copy16(const void* g, void* l) {
#if HAVE_ASYNC_B128
  __builtin_amdgcn_global_load_async_to_lds_b128(
      (AS1 v4i*)g, (AS3 v4i*)l, 0, 0);
#else
  *(uint4*)l = *(const uint4*)g;
#endif
}

__device__ __forceinline__ void wait_async0() {
#if HAVE_ASYNC_B128
#  if HAVE_WAIT_ASYNC
  __builtin_amdgcn_s_wait_asynccnt(0);
#  else
  asm volatile("s_wait_asynccnt 0x0" ::: "memory");
#  endif
#endif
}

__device__ __forceinline__ void wait_tensor0() {
#if HAVE_WAIT_TENSOR
  __builtin_amdgcn_s_wait_tensorcnt(0);
#else
  asm volatile("s_wait_tensorcnt 0x0" ::: "memory");
#endif
}

#if HAVE_TDM
typedef __attribute__((ext_vector_type(4))) unsigned int v4u_t;
typedef __attribute__((ext_vector_type(8))) int v8i_t;
typedef __attribute__((ext_vector_type(4))) int v4i_t;

// TDM: DMA a 32x64 bf16 tile (global row stride 64 elem) into LDS rows padded
// to 80 elem (pad_interval = 32 DWORDs -> one 64-elem row; pad_amount = 8
// DWORDs -> 16 elems) so LDS layout matches Ks[32][80].
__device__ __forceinline__ void tdm_load_k32x64(const __bf16* gsrc,
                                                unsigned lds_byte_off) {
  unsigned long long ga = (unsigned long long)(uintptr_t)gsrc;
  v4u_t g0;
  g0[0] = 1u;                                        // count=1, user mode
  g0[1] = lds_byte_off;                              // lds_addr
  g0[2] = (unsigned)(ga & 0xffffffffu);              // global_addr[31:0]
  g0[3] = (unsigned)((ga >> 32) & 0x1ffffffu)        // global_addr[56:32]
          | (2u << 30);                              // type = 2 ("image")
  v8i_t g1;
  g1[0] = (int)((1u << 16)      // data_size = 1 (2 bytes)
                | (1u << 20)    // pad_enable
                | (4u << 22)    // pad_interval: 32 DWORDs
                | (7u << 25));  // pad_amount:   8 DWORDs
  g1[1] = (int)(64u << 16);     // tensor_dim0 = 64 (low16 in [63:48])
  g1[2] = (int)(32u << 16);     // tensor_dim1 = 32 (low16 in [95:80])
  g1[3] = (int)(64u << 16);     // tile_dim0 = 64   ([127:112])
  g1[4] = 32;                   // tile_dim1 = 32   ([143:128]); tile_dim2=0
  g1[5] = 64;                   // tensor_dim0_stride = 64 ([191:160])
  g1[6] = 0;
  g1[7] = 0;
  v4i_t z4 = {0, 0, 0, 0};
#if __clang_major__ >= 23
  v8i_t z8 = {0, 0, 0, 0, 0, 0, 0, 0};
  __builtin_amdgcn_tensor_load_to_lds(g0, g1, z4, z4, z8, 0);
#else
  __builtin_amdgcn_tensor_load_to_lds(g0, g1, z4, z4, 0);
#endif
}
#endif  // HAVE_TDM

// ---------------------------------------------------------------------------
// fp32 -> bf16 conversion
// ---------------------------------------------------------------------------
__global__ void cvt_f32_bf16(const float* __restrict__ src,
                             __bf16* __restrict__ dst, int n) {
  int i = blockIdx.x * blockDim.x + threadIdx.x;
  if (i < n) dst[i] = (__bf16)src[i];
}

// ---------------------------------------------------------------------------
// Tiled WMMA GEMM:  Out[M,N] = A[M,K] @ W[N,K]^T   (both row-major, K contig)
// Block tile 128x128, BK=32, 256 threads = 8 waves, each wave -> 64x32 region
// MODE 0: Out row-major [M,N]
// MODE 1: scatter Out into [B, NHEAD, SEQ, HDIM]   (m=(b,t), n=(h,d))
// ---------------------------------------------------------------------------
template <typename OutT, int MODE>
__global__ __launch_bounds__(256)
void gemm_wmma(const __bf16* __restrict__ A, const __bf16* __restrict__ W,
               OutT* __restrict__ Out, int M, int N, int K) {
  __shared__ __bf16 As[128][48];   // 32 cols used, padded to 48 (96B rows)
  __shared__ __bf16 Ws[128][48];

  const int tid  = threadIdx.x;
  const int lane = tid & 31;
  const int wid  = tid >> 5;
  const int wm   = wid >> 2;            // 0..1  (64 rows each)
  const int wn   = wid & 3;             // 0..3  (32 cols each)
  const int bm   = blockIdx.x * 128;
  const int bn   = blockIdx.y * 128;

  const int rl   = lane & 15;
  const int kb   = (lane >> 4) * 16;    // K-half select
  const int half = lane >> 4;

  v8f acc[4][2] = {};

  const int arow = tid >> 1;            // 0..127
  const int acol = (tid & 1) * 16;      // 0 or 16

  for (int k0 = 0; k0 < K; k0 += 32) {
    // stage A / W tiles (each thread: one 32-byte chunk per tile)
    async_copy16(&A[(size_t)(bm + arow) * K + k0 + acol], &As[arow][acol]);
    async_copy16(&A[(size_t)(bm + arow) * K + k0 + acol + 8],
                 &As[arow][acol + 8]);
    async_copy16(&W[(size_t)(bn + arow) * K + k0 + acol], &Ws[arow][acol]);
    async_copy16(&W[(size_t)(bn + arow) * K + k0 + acol + 8],
                 &Ws[arow][acol + 8]);
    wait_async0();
    __syncthreads();

    v16bf bfrag[2];
#pragma unroll
    for (int j = 0; j < 2; ++j)
      bfrag[j] = *(const v16bf*)&Ws[wn * 32 + j * 16 + rl][kb];

#pragma unroll
    for (int i = 0; i < 4; ++i) {
      v16bf afrag = *(const v16bf*)&As[wm * 64 + i * 16 + rl][kb];
#pragma unroll
      for (int j = 0; j < 2; ++j)
        acc[i][j] = __builtin_amdgcn_wmma_f32_16x16x32_bf16(
            false, afrag, false, bfrag[j], (short)0, acc[i][j], false, false);
    }
    __syncthreads();
  }

  // Epilogue: C/D layout => lane holds fixed column n, rows r + 8*half
#pragma unroll
  for (int i = 0; i < 4; ++i)
#pragma unroll
    for (int j = 0; j < 2; ++j) {
      int n = bn + wn * 32 + j * 16 + rl;
#pragma unroll
      for (int r = 0; r < 8; ++r) {
        int m = bm + wm * 64 + i * 16 + r + 8 * half;
        float v = acc[i][j][r];
        if (MODE == 0) {
          Out[(size_t)m * N + n] = (OutT)v;
        } else {
          int b = m / SEQ, t = m % SEQ;
          int h = n / HDIM, d = n % HDIM;
          Out[(((size_t)b * NHEAD + h) * SEQ + t) * HDIM + d] = (OutT)v;
        }
      }
    }
}

// ---------------------------------------------------------------------------
// RoPE applied in place to Q and K, both [B, NHEAD, SEQ, HDIM] bf16
// ---------------------------------------------------------------------------
__global__ void rope_kernel(__bf16* __restrict__ q, __bf16* __restrict__ k) {
  int idx = blockIdx.x * blockDim.x + threadIdx.x;   // over B*H*T*(HDIM/2)
  const int total = BATCH * NHEAD * SEQ * (HDIM / 2);
  if (idx >= total) return;
  int j = idx & 31;                    // freq index 0..31
  int row = idx >> 5;                  // (b,h,t) flattened
  int t = row % SEQ;
  size_t base = (size_t)row * HDIM;

  float inv = __expf(-((float)(2 * j) / (float)HDIM) * __logf(10000.0f));
  float ang = (float)t * inv;
  float c = __cosf(ang), s = __sinf(ang);

  float q1 = (float)q[base + j], q2 = (float)q[base + j + 32];
  q[base + j]      = (__bf16)(q1 * c - q2 * s);
  q[base + j + 32] = (__bf16)(q2 * c + q1 * s);

  float k1 = (float)k[base + j], k2 = (float)k[base + j + 32];
  k[base + j]      = (__bf16)(k1 * c - k2 * s);
  k[base + j + 32] = (__bf16)(k2 * c + k1 * s);
}

// ---------------------------------------------------------------------------
// Flash attention: one block = one (b,h) x 128 q-rows. 8 waves, each wave owns
// a 16-row Q tile. Keys streamed in chunks of 32 through LDS (TDM / async
// when available); online softmax with wave-uniform causal specialization.
// Y written as [B, SEQ, CH] bf16 (pre-Wo).
// ---------------------------------------------------------------------------
__global__ __launch_bounds__(256)
void attn_wmma(const __bf16* __restrict__ Q, const __bf16* __restrict__ K,
               const __bf16* __restrict__ V, __bf16* __restrict__ Y) {
  __shared__ __bf16 Ks[32][80];       // 32 keys x 64 d (padded)
  __shared__ __bf16 VT[64][48];       // transposed: 64 d x 32 keys (padded)
  __shared__ __bf16 Ps[8][16][48];    // per-wave P tile 16x32 (padded)

  const int tid  = threadIdx.x;
  const int lane = tid & 31;
  const int wid  = tid >> 5;
  const int rl   = lane & 15;
  const int half = lane >> 4;
  const int kb   = half * 16;

  const int bh = blockIdx.y;                 // b*NHEAD + h
  const int qb = blockIdx.x * 128;
  const size_t head_off = (size_t)bh * SEQ * HDIM;
  const __bf16* Qh = Q + head_off;
  const __bf16* Kh = K + head_off;
  const __bf16* Vh = V + head_off;

  const int qrow0 = qb + wid * 16;           // this wave's q rows

  // Q fragments: 16x32 (d 0..31) and 16x32 (d 32..63).
  // Fold the softmax scale 1/sqrt(64)=0.125 into Q once (exact in bf16).
  v16bf qf0 = *(const v16bf*)&Qh[(size_t)(qrow0 + rl) * HDIM + kb];
  v16bf qf1 = *(const v16bf*)&Qh[(size_t)(qrow0 + rl) * HDIM + 32 + kb];
#pragma unroll
  for (int e = 0; e < 16; ++e) {
    qf0[e] = (__bf16)((float)qf0[e] * 0.125f);
    qf1[e] = (__bf16)((float)qf1[e] * 0.125f);
  }

  v8f o[4] = {};                             // 4 d-groups of 16 columns
  float mrow[8], lrow[8];
#pragma unroll
  for (int r = 0; r < 8; ++r) { mrow[r] = -INFINITY; lrow[r] = 0.0f; }

#if HAVE_TDM
  const unsigned ks_lds_off =
      (unsigned)(uintptr_t)(AS3 char*)&Ks[0][0];
#endif

  const int kend = qb + 128;                 // causal: keys [0, kend)
  for (int kc = 0; kc < kend; kc += 32) {
    // ---- stage K chunk (row-major) and V chunk (transposed) ----
#if HAVE_TDM
    if (tid == 0) {
      tdm_load_k32x64(&Kh[(size_t)kc * HDIM], ks_lds_off);
      wait_tensor0();
    }
#else
    {
      int row = tid >> 3;                    // 0..31
      int col = (tid & 7) * 8;               // 0..56
      async_copy16(&Kh[(size_t)(kc + row) * HDIM + col], &Ks[row][col]);
      if (kc + 32 < kend)
        __builtin_prefetch(&Kh[(size_t)(kc + 32 + row) * HDIM + col], 0, 1);
    }
#endif
    {
#pragma unroll
      for (int e = 0; e < 8; ++e) {
        int idx = tid * 8 + e;               // 0..2047
        int key = idx >> 6, d = idx & 63;
        VT[d][key] = Vh[(size_t)(kc + key) * HDIM + d];
      }
    }
    wait_async0();
    __syncthreads();

    // Wave-uniform causal classification (EXEC stays all-1s for WMMA):
    //   kc + 31 <= qrow0          -> fully visible, no masking needed
    //   kc      <= qrow0 + 15     -> diagonal, per-element mask
    //   otherwise                 -> fully masked, skip all compute
    if (kc <= qrow0 + 15) {
      const bool full = (kc + 31 <= qrow0);

      // ---- S = Q @ K^T for two 16-key tiles ----
      v8f s[2];
#pragma unroll
      for (int kt = 0; kt < 2; ++kt) {
        v16bf kf0 = *(const v16bf*)&Ks[kt * 16 + rl][kb];        // d 0..31
        v16bf kf1 = *(const v16bf*)&Ks[kt * 16 + rl][32 + kb];   // d 32..63
        v8f z = {};
        z = __builtin_amdgcn_wmma_f32_16x16x32_bf16(false, qf0, false, kf0,
                                                    (short)0, z, false, false);
        z = __builtin_amdgcn_wmma_f32_16x16x32_bf16(false, qf1, false, kf1,
                                                    (short)0, z, false, false);
        s[kt] = z;
      }

      // ---- online softmax over this 32-key chunk ----
#pragma unroll
      for (int r = 0; r < 8; ++r) {
        float v0 = s[0][r];
        float v1 = s[1][r];
        if (!full) {
          int qrow = qrow0 + r + 8 * half;
          v0 = (kc + rl <= qrow) ? v0 : -INFINITY;
          v1 = (kc + 16 + rl <= qrow) ? v1 : -INFINITY;
        }

        float m = fmaxf(v0, v1);
#pragma unroll
        for (int off = 1; off < 16; off <<= 1)
          m = fmaxf(m, __shfl_xor(m, off, 32));
        float mnew = fmaxf(mrow[r], m);

        float p0 = __expf(v0 - mnew);
        float p1 = __expf(v1 - mnew);
        float rs = p0 + p1;
#pragma unroll
        for (int off = 1; off < 16; off <<= 1)
          rs += __shfl_xor(rs, off, 32);

        float alpha = __expf(mrow[r] - mnew);  // 0 on first chunk
        lrow[r] = lrow[r] * alpha + rs;
        mrow[r] = mnew;
#pragma unroll
        for (int g = 0; g < 4; ++g) o[g][r] *= alpha;

        // scatter P (C/D layout -> row-major LDS tile) as bf16
        Ps[wid][r + 8 * half][rl]      = (__bf16)p0;
        Ps[wid][r + 8 * half][16 + rl] = (__bf16)p1;
      }
      asm volatile("s_wait_dscnt 0x0" ::: "memory");

      // ---- O += P @ V ----
      v16bf pf = *(const v16bf*)&Ps[wid][rl][kb];   // A frag 16x32 (keys)
#pragma unroll
      for (int g = 0; g < 4; ++g) {
        v16bf vf = *(const v16bf*)&VT[g * 16 + rl][kb];  // B: lane=d, 16 keys
        o[g] = __builtin_amdgcn_wmma_f32_16x16x32_bf16(false, pf, false, vf,
                                                       (short)0, o[g], false,
                                                       false);
      }
    }
    __syncthreads();
  }

  // ---- normalize and write Y [B, SEQ, CH] ----
  const int b = bh >> 4;       // / NHEAD
  const int h = bh & 15;       // % NHEAD
#pragma unroll
  for (int g = 0; g < 4; ++g)
#pragma unroll
    for (int r = 0; r < 8; ++r) {
      int t = qrow0 + r + 8 * half;
      int c = h * HDIM + g * 16 + rl;
      float val = o[g][r] / lrow[r];
      Y[((size_t)b * SEQ + t) * CH + c] = (__bf16)val;
    }
}

// ---------------------------------------------------------------------------
// Host-side orchestration
// ---------------------------------------------------------------------------
static inline size_t align256(size_t x) { return (x + 255) & ~(size_t)255; }

extern "C" void kernel_launch(void* const* d_in, const int* in_sizes, int n_in,
                              void* d_out, int out_size, void* d_ws,
                              size_t ws_size, hipStream_t stream) {
  const float* x  = (const float*)d_in[0];
  const float* Wq = (const float*)d_in[1];
  const float* Wk = (const float*)d_in[2];
  const float* Wv = (const float*)d_in[3];
  const float* Wo = (const float*)d_in[4];
  float* out = (float*)d_out;

  const size_t xN = (size_t)MROWS * CH;   // 4M elems
  const size_t wN = (size_t)CH * CH;      // 1M elems

  char* ws = (char*)d_ws;
  size_t off = 0;
  __bf16* xb = (__bf16*)(ws + off); off = align256(off + xN * 2);
  __bf16* wq = (__bf16*)(ws + off); off = align256(off + wN * 2);
  __bf16* wk = (__bf16*)(ws + off); off = align256(off + wN * 2);
  __bf16* wv = (__bf16*)(ws + off); off = align256(off + wN * 2);
  __bf16* wo = (__bf16*)(ws + off); off = align256(off + wN * 2);
  __bf16* qb = (__bf16*)(ws + off); off = align256(off + xN * 2);
  __bf16* kb = (__bf16*)(ws + off); off = align256(off + xN * 2);
  __bf16* vb = (__bf16*)(ws + off); off = align256(off + xN * 2);
  __bf16* yb = (__bf16*)(ws + off); off = align256(off + xN * 2);
  (void)ws_size; (void)n_in; (void)in_sizes; (void)out_size;

  // 1. convert to bf16
  cvt_f32_bf16<<<(int)((xN + 255) / 256), 256, 0, stream>>>(x, xb, (int)xN);
  cvt_f32_bf16<<<(int)((wN + 255) / 256), 256, 0, stream>>>(Wq, wq, (int)wN);
  cvt_f32_bf16<<<(int)((wN + 255) / 256), 256, 0, stream>>>(Wk, wk, (int)wN);
  cvt_f32_bf16<<<(int)((wN + 255) / 256), 256, 0, stream>>>(Wv, wv, (int)wN);
  cvt_f32_bf16<<<(int)((wN + 255) / 256), 256, 0, stream>>>(Wo, wo, (int)wN);

  // 2. Q/K/V projections: [M, C] @ W^T, scattered into [B, H, T, D]
  dim3 ggrid(MROWS / 128, CH / 128);
  gemm_wmma<__bf16, 1><<<ggrid, 256, 0, stream>>>(xb, wq, qb, MROWS, CH, CH);
  gemm_wmma<__bf16, 1><<<ggrid, 256, 0, stream>>>(xb, wk, kb, MROWS, CH, CH);
  gemm_wmma<__bf16, 1><<<ggrid, 256, 0, stream>>>(xb, wv, vb, MROWS, CH, CH);

  // 3. RoPE on Q, K
  const int rope_total = BATCH * NHEAD * SEQ * (HDIM / 2);
  rope_kernel<<<(rope_total + 255) / 256, 256, 0, stream>>>(qb, kb);

  // 4. causal flash attention -> Y [B, T, C] bf16
  dim3 agrid(SEQ / 128, BATCH * NHEAD);
  attn_wmma<<<agrid, 256, 0, stream>>>(qb, kb, vb, yb);

  // 5. output projection: out = Y @ Wo^T  (fp32)
  gemm_wmma<float, 0><<<ggrid, 256, 0, stream>>>(yb, wo, out, MROWS, CH, CH);
}
